// AttentionHead_670014898299
// MI455X (gfx1250) — matrix operand order repaired and checked by reading, not compiled, not run on hardware
//
#include <hip/hip_runtime.h>
#include <hip/hip_bf16.h>

#define T_DIM 8192
#define C_DIM 1024
#define H_DIM 64
#define SCALE 0.125f  // 64^-0.5, folded into Q at projection time
#define LDS_PAD 72    // 144B row stride: conflict-free 16-lane fragment reads

typedef __bf16 bf16_t;
typedef __attribute__((ext_vector_type(8)))  __bf16 v8bf;
typedef __attribute__((ext_vector_type(16))) __bf16 v16bf;
typedef __attribute__((ext_vector_type(8)))  float  v8f;

__device__ __forceinline__ bf16_t f2bf(float f) { return (bf16_t)f; }

// ---- WMMA fragment loaders (wave32, 16x16x32 bf16) -------------------------
// A: 16x32 (MxK). lane L: m = L%16, g = L/16. elements j<8 -> k0+8g+j,
// j>=8 -> k0+16+8g+(j-8). Two contiguous 16B loads from row-major [m][k].
__device__ __forceinline__ v16bf load_a16(const bf16_t* __restrict__ base,
                                          int ldm, int m, int g, int k0) {
  const bf16_t* p = base + (size_t)m * ldm + k0 + 8 * g;
  v8bf lo = *(const v8bf*)(p);
  v8bf hi = *(const v8bf*)(p + 16);
  return __builtin_shufflevector(lo, hi, 0,1,2,3,4,5,6,7,8,9,10,11,12,13,14,15);
}

// B: 32x16 (KxN), B[k][n] = src[row=n][k] of a row-major source.
// lane L: n = L%16, g = L/16. elements j<8 -> k0+8g+j,
// j>=8 -> k0+16+8g+(j-8).
__device__ __forceinline__ v16bf load_b16(const bf16_t* __restrict__ base,
                                          int ldm, int row, int g, int k0) {
  const bf16_t* p = base + (size_t)row * ldm + k0 + 8 * g;
  v8bf lo = *(const v8bf*)(p);
  v8bf hi = *(const v8bf*)(p + 16);
  return __builtin_shufflevector(lo, hi, 0,1,2,3,4,5,6,7,8,9,10,11,12,13,14,15);
}

// reuse_a: legal when the previous matrix instruction is the same WMMA with
// the same A registers -> HW keeps A in the operand cache (ISA 7.12 RA hint).
template <bool REUSE_A>
__device__ __forceinline__ v8f wmma_bf16(v16bf a, v16bf b, v8f c) {
  return __builtin_amdgcn_wmma_f32_16x16x32_bf16(
      false, a, false, b, (short)0, c, REUSE_A, false);
}

// ---- async copy: global -> LDS, 16B per lane, ASYNCcnt-tracked -------------
__device__ __forceinline__ void async_b128(uint32_t lds_off,
                                           const bf16_t* gptr) {
  asm volatile("global_load_async_to_lds_b128 %0, %1, off"
               :: "v"(lds_off), "v"(gptr)
               : "memory");
}

// ---- k0: f32 -> bf16 conversion of x and [Wk|Wq|Wv] ------------------------
__global__ void __launch_bounds__(256) k0_convert(
    const float* __restrict__ x, const float* __restrict__ Wk,
    const float* __restrict__ Wq, const float* __restrict__ Wv,
    bf16_t* __restrict__ xb, bf16_t* __restrict__ wb) {
  size_t i = (size_t)blockIdx.x * 256 + threadIdx.x;
  const size_t NX = (size_t)T_DIM * C_DIM;
  const size_t NW = (size_t)H_DIM * C_DIM;
  if (i < NX) {
    xb[i] = f2bf(x[i]);
  } else {
    size_t j = i - NX;
    if (j < 3 * NW) {
      size_t sel = j / NW, off = j % NW;
      const float* s = (sel == 0) ? Wk : (sel == 1) ? Wq : Wv;
      wb[j] = f2bf(s[off]);
    }
  }
}

// ---- k1: projections, software-pipelined in groups of 6 B-fragments --------
// Outputs: kbm,qbm row-major (T x 64), vtb transposed (64 x T).
// Q is pre-scaled by 2^-3 (exact in bf16).
__global__ void __launch_bounds__(32) k1_project(
    const bf16_t* __restrict__ xb, const bf16_t* __restrict__ wb,
    bf16_t* __restrict__ kbm, bf16_t* __restrict__ qbm,
    bf16_t* __restrict__ vtb) {
  const int lane = threadIdx.x & 15;
  const int g    = (threadIdx.x >> 4) & 1;
  const int mb   = blockIdx.x * 16;

  v8f z = {0.f,0.f,0.f,0.f,0.f,0.f,0.f,0.f};
  v8f acc[12];
#pragma unroll
  for (int t = 0; t < 12; ++t) acc[t] = z;

  const bf16_t* xrow = xb + (size_t)mb * C_DIM;
  for (int k0 = 0; k0 < C_DIM; k0 += 32) {
    v16bf a = load_a16(xrow, C_DIM, lane, g, k0);
#pragma unroll
    for (int grp = 0; grp < 2; ++grp) {
      v16bf b[6];
#pragma unroll
      for (int u = 0; u < 6; ++u) {
        int t = grp * 6 + u;
        const bf16_t* wsrc = wb + (size_t)(t >> 2) * H_DIM * C_DIM;
        b[u] = load_b16(wsrc, C_DIM, (t & 3) * 16 + lane, g, k0);
      }
      // same A across all 12 WMMAs of this k-step -> reuse_a after the first
      if (grp == 0) {
        acc[0] = wmma_bf16<false>(a, b[0], acc[0]);
#pragma unroll
        for (int u = 1; u < 6; ++u)
          acc[u] = wmma_bf16<true>(a, b[u], acc[u]);
      } else {
#pragma unroll
        for (int u = 0; u < 6; ++u)
          acc[6 + u] = wmma_bf16<true>(a, b[u], acc[6 + u]);
      }
    }
  }

#pragma unroll
  for (int t = 0; t < 12; ++t) {
    int n = (t & 3) * 16 + lane;
#pragma unroll
    for (int r = 0; r < 8; ++r) {
      int row = mb + r + 8 * g;
      if (t < 4)       kbm[(size_t)row * H_DIM + n] = f2bf(acc[t][r]);
      else if (t < 8)  qbm[(size_t)row * H_DIM + n] = f2bf(acc[t][r] * SCALE);
      else             vtb[(size_t)n * T_DIM + row] = f2bf(acc[t][r]);
    }
  }
}

// ---- k2: obj_affinities = q_scaled[1] . k_j  (scale already folded in) -----
__global__ void __launch_bounds__(256) k2_obj(
    const bf16_t* __restrict__ qbm, const bf16_t* __restrict__ kbm,
    float* __restrict__ obj) {
  int j = blockIdx.x * 256 + threadIdx.x;
  float acc = 0.f;
#pragma unroll
  for (int d = 0; d < H_DIM; ++d)
    acc += (float)qbm[H_DIM + d] * (float)kbm[(size_t)j * H_DIM + d];
  obj[j] = acc;
}

// ---- k3: flash attention, double-buffered async-to-LDS K/V staging ---------
// 4 waves/block share each 64-key chunk; chunk i+1 streams in via the async
// engine while chunk i is consumed from LDS.
__global__ void __launch_bounds__(128) k3_attn(
    const bf16_t* __restrict__ qbm, const bf16_t* __restrict__ kbm,
    const bf16_t* __restrict__ vtb, float* __restrict__ out) {
  __shared__ __align__(16) bf16_t kbuf[2][64 * LDS_PAD];
  __shared__ __align__(16) bf16_t vbuf[2][64 * LDS_PAD];
  __shared__ __align__(16) bf16_t pbuf[4][16 * LDS_PAD];

  const int tid  = threadIdx.x;
  const int w    = tid >> 5;
  const int lane = tid & 15;
  const int g    = (tid >> 4) & 1;
  const int mb   = (blockIdx.x * 4 + w) * 16;
  bf16_t* pb = pbuf[w];

  // Q fragments for this M tile (pre-scaled), reused across all chunks.
  v16bf aq0 = load_a16(qbm + (size_t)mb * H_DIM, H_DIM, lane, g, 0);
  v16bf aq1 = load_a16(qbm + (size_t)mb * H_DIM, H_DIM, lane, g, 32);

  v8f z = {0.f,0.f,0.f,0.f,0.f,0.f,0.f,0.f};
  v8f o[4];
#pragma unroll
  for (int n = 0; n < 4; ++n) o[n] = z;
  float m_i[8], l_i[8];
#pragma unroll
  for (int r = 0; r < 8; ++r) { m_i[r] = -3.0e38f; l_i[r] = 0.f; }

  // stage(buf, kb0): 512 16B pieces for K, 512 for V; 8 async ops per wave.
  auto stage = [&](int buf, int kb0) {
#pragma unroll
    for (int p = 0; p < 4; ++p) {
      int idx = tid + p * 128;                 // 0..511
      int row = idx >> 3, seg = idx & 7;       // row in chunk, 16B segment
      async_b128((uint32_t)(uintptr_t)(kbuf[buf] + row * LDS_PAD + seg * 8),
                 kbm + (size_t)(kb0 + row) * H_DIM + seg * 8);
    }
#pragma unroll
    for (int p = 0; p < 4; ++p) {
      int idx = tid + p * 128;
      int row = idx >> 3, seg = idx & 7;       // row = head dim, seg over keys
      async_b128((uint32_t)(uintptr_t)(vbuf[buf] + row * LDS_PAD + seg * 8),
                 vtb + (size_t)row * T_DIM + kb0 + seg * 8);
    }
  };

  const int NCHUNK = T_DIM / 64;
  stage(0, 0);

  for (int c = 0; c < NCHUNK; ++c) {
    const int cur = c & 1;
    if (c + 1 < NCHUNK) {
      stage(cur ^ 1, (c + 1) * 64);
      asm volatile("s_wait_asynccnt 8" ::: "memory");   // chunk c landed
    } else {
      asm volatile("s_wait_asynccnt 0" ::: "memory");
    }
    __syncthreads();
    const bf16_t* kc = kbuf[cur];
    const bf16_t* vc = vbuf[cur];

    // ---- S = Qs K^T for 64 keys: load all 8 K fragments, then 2 WMMA ------
    // passes (all-aq0 then all-aq1) so the A-reuse chain applies.
    v16bf bk0[4], bk1[4];
#pragma unroll
    for (int t = 0; t < 4; ++t) {
      int krow = t * 16 + lane;
      bk0[t] = load_b16(kc, LDS_PAD, krow, g, 0);
      bk1[t] = load_b16(kc, LDS_PAD, krow, g, 32);
    }
    v8f s[4];
    s[0] = wmma_bf16<false>(aq0, bk0[0], z);
#pragma unroll
    for (int t = 1; t < 4; ++t) s[t] = wmma_bf16<true>(aq0, bk0[t], z);
    s[0] = wmma_bf16<false>(aq1, bk1[0], s[0]);
#pragma unroll
    for (int t = 1; t < 4; ++t) s[t] = wmma_bf16<true>(aq1, bk1[t], s[t]);

    // ---- online softmax update (rows live across 16-lane halves) ----------
#pragma unroll
    for (int r = 0; r < 8; ++r) {
      float cm = fmaxf(fmaxf(s[0][r], s[1][r]), fmaxf(s[2][r], s[3][r]));
      cm = fmaxf(cm, __shfl_xor(cm, 1, 32));
      cm = fmaxf(cm, __shfl_xor(cm, 2, 32));
      cm = fmaxf(cm, __shfl_xor(cm, 4, 32));
      cm = fmaxf(cm, __shfl_xor(cm, 8, 32));
      float nm    = fmaxf(m_i[r], cm);
      float alpha = __expf(m_i[r] - nm);
      m_i[r] = nm;
      float rs = 0.f;
#pragma unroll
      for (int t = 0; t < 4; ++t) {
        float pv = __expf(s[t][r] - nm);
        s[t][r] = pv;
        rs += pv;
      }
      rs += __shfl_xor(rs, 1, 32);
      rs += __shfl_xor(rs, 2, 32);
      rs += __shfl_xor(rs, 4, 32);
      rs += __shfl_xor(rs, 8, 32);
      l_i[r] = l_i[r] * alpha + rs;
#pragma unroll
      for (int n = 0; n < 4; ++n) o[n][r] *= alpha;
    }

    // ---- P: C-layout -> row-major LDS (bf16) -> A-layout ------------------
#pragma unroll
    for (int t = 0; t < 4; ++t)
#pragma unroll
      for (int r = 0; r < 8; ++r)
        pb[(r + 8 * g) * LDS_PAD + t * 16 + lane] = f2bf(s[t][r]);
    __syncthreads();

    // ---- O += P @ V^T: load A + all 4 V fragments, then WMMA group with ---
    // A-reuse (same P fragment feeds 4 consecutive WMMAs).
#pragma unroll
    for (int ks = 0; ks < 2; ++ks) {
      v16bf ap = load_a16(pb, LDS_PAD, lane, g, ks * 32);
      v16bf bv[4];
#pragma unroll
      for (int n = 0; n < 4; ++n)
        bv[n] = load_b16(vc, LDS_PAD, n * 16 + lane, g, ks * 32);
      o[0] = wmma_bf16<false>(ap, bv[0], o[0]);
#pragma unroll
      for (int n = 1; n < 4; ++n) o[n] = wmma_bf16<true>(ap, bv[n], o[n]);
    }
    __syncthreads();   // pbuf reuse + buffer (cur) free for restage
  }

  // ---- normalize and write out (row-major T x 64, f32) ---------------------
#pragma unroll
  for (int n = 0; n < 4; ++n)
#pragma unroll
    for (int r = 0; r < 8; ++r) {
      int row = mb + r + 8 * g;
      out[(size_t)row * H_DIM + n * 16 + lane] = o[n][r] / l_i[r];
    }
}

extern "C" void kernel_launch(void* const* d_in, const int* in_sizes, int n_in,
                              void* d_out, int out_size, void* d_ws, size_t ws_size,
                              hipStream_t stream) {
  const float* x  = (const float*)d_in[0];
  const float* Wk = (const float*)d_in[1];
  const float* Wq = (const float*)d_in[2];
  const float* Wv = (const float*)d_in[3];

  float* outp = (float*)d_out;                  // (8192 x 64) f32
  float* objp = outp + (size_t)T_DIM * H_DIM;   // (8192) f32

  char* ws = (char*)d_ws;
  const size_t XB_B = (size_t)T_DIM * C_DIM * 2;        // 16 MB
  const size_t WB_B = (size_t)3 * H_DIM * C_DIM * 2;    // 384 KB
  const size_t QK_B = (size_t)T_DIM * H_DIM * 2;        // 1 MB each
  bf16_t* xb  = (bf16_t*)ws;
  bf16_t* wb  = (bf16_t*)(ws + XB_B);
  bf16_t* qbm = (bf16_t*)(ws + XB_B + WB_B);
  bf16_t* kbm = (bf16_t*)(ws + XB_B + WB_B + QK_B);
  bf16_t* vtb = (bf16_t*)(ws + XB_B + WB_B + 2 * QK_B);

  const size_t total = (size_t)T_DIM * C_DIM + (size_t)3 * H_DIM * C_DIM;
  int cvt_blocks = (int)((total + 255) / 256);

  k0_convert<<<cvt_blocks, 256, 0, stream>>>(x, Wk, Wq, Wv, xb, wb);
  k1_project<<<T_DIM / 16, 32, 0, stream>>>(xb, wb, kbm, qbm, vtb);
  k2_obj<<<T_DIM / 256, 256, 0, stream>>>(qbm, kbm, objp);
  k3_attn<<<T_DIM / 64, 128, 0, stream>>>(qbm, kbm, vtb, outp);
}
